// TopKSAEBackend_79998060855606
// MI455X (gfx1250) — compile-verified, matrix operand order checked by reading, not run on hardware
//
#include <hip/hip_runtime.h>

typedef float  v8f  __attribute__((ext_vector_type(8)));
typedef __bf16 v4bf __attribute__((ext_vector_type(4)));
typedef __bf16 v8bf __attribute__((ext_vector_type(8)));
typedef __bf16 v16bf __attribute__((ext_vector_type(16)));

#define D_MODEL 2048
#define D_SAE   16384
#define K_TOP   64
#define N_TOK   8192

#define BTM 256      // block tile rows  (8 waves x 32 rows)
#define BTN 128      // block tile cols
#define KC  32       // k chunk == WMMA K depth
#define LDP 40       // padded LDS row stride (bf16 elems): 80B, 16B-aligned,
                     // row-to-row bank step of 20 dwords -> conflict-free b128 frags

// ---------------------------------------------------------------------------
// f32 -> (hi, lo) bf16 split:  f = hi + lo + O(2^-16 * lo)
// ---------------------------------------------------------------------------
__device__ __forceinline__ __bf16 bf_hi(float f) { return (__bf16)f; }
__device__ __forceinline__ __bf16 bf_lo(float f, __bf16 hi) {
  return (__bf16)(f - (float)hi);
}

// Fragment per ISA 16-bit A/B layout: lane elems [0..7] = K lhalf*8..+7,
// elems [8..15] = K 16+lhalf*8..+7  -> two 16B LDS loads.
__device__ __forceinline__ v16bf load_frag(const __bf16* rowbase, int lhalf) {
  v8bf p0 = *(const v8bf*)(rowbase + lhalf * 8);
  v8bf p1 = *(const v8bf*)(rowbase + 16 + lhalf * 8);
  return __builtin_shufflevector(p0, p1, 0, 1, 2, 3, 4, 5, 6, 7,
                                         8, 9, 10, 11, 12, 13, 14, 15);
}

// ---------------------------------------------------------------------------
// Kernel 1: pre = (x - b_dec) @ W_enc + b_enc
// bf16x3 split-precision emulation on V_WMMA_F32_16X16X32_BF16.
// Block: 256 threads (8 wave32); tile 256 rows x 128 cols; wave owns 32 rows.
// ---------------------------------------------------------------------------
__global__ __launch_bounds__(256) void sae_encode_gemm(
    const float* __restrict__ x, const float* __restrict__ W_enc,
    const float* __restrict__ b_enc, const float* __restrict__ b_dec,
    float* __restrict__ pre)
{
  __shared__ __bf16 lAh[BTM][LDP];   // (x - b_dec) hi, [n][k]
  __shared__ __bf16 lAl[BTM][LDP];   // (x - b_dec) lo
  __shared__ __bf16 lBh[BTN][LDP];   // W_enc hi, transposed [m][k]
  __shared__ __bf16 lBl[BTN][LDP];   // W_enc lo

  const int tid   = threadIdx.x;
  const int lane  = tid & 31;
  const int wv    = tid >> 5;          // wave id 0..7
  const int row0  = blockIdx.y * BTM;
  const int col0  = blockIdx.x * BTN;
  const int lhalf = lane >> 4;         // selects K-half inside fragments
  const int l15   = lane & 15;         // M (A) / N (B, C, D)

  // Accumulators (16 x 16x16 tiles) start at b_enc (column-only bias)
  v8f acc[8][2];
  #pragma unroll
  for (int t = 0; t < 8; ++t) {
    float be = b_enc[col0 + t * 16 + l15];
    v8f a = {be, be, be, be, be, be, be, be};
    acc[t][0] = a;
    acc[t][1] = a;
  }

  for (int k0 = 0; k0 < D_MODEL; k0 += KC) {
    // ---- stage A: 256x32 floats; subtract b_dec, split hi/lo. 8 float4/thread
    #pragma unroll
    for (int j = 0; j < 8; ++j) {
      int li = tid + j * 256;          // 0..2047 float4 slots
      int n  = li >> 3;                // 8 float4 per 32-float row
      int kc = li & 7;
      float4 xv = *(const float4*)&x[(size_t)(row0 + n) * D_MODEL + k0 + kc * 4];
      float4 bd = *(const float4*)&b_dec[k0 + kc * 4];
      float f0 = xv.x - bd.x, f1 = xv.y - bd.y, f2 = xv.z - bd.z, f3 = xv.w - bd.w;
      __bf16 h0 = bf_hi(f0), h1 = bf_hi(f1), h2 = bf_hi(f2), h3 = bf_hi(f3);
      __bf16 l0 = bf_lo(f0, h0), l1 = bf_lo(f1, h1);
      __bf16 l2 = bf_lo(f2, h2), l3 = bf_lo(f3, h3);
      v4bf h = {h0, h1, h2, h3};
      v4bf l = {l0, l1, l2, l3};
      *(v4bf*)&lAh[n][kc * 4] = h;
      *(v4bf*)&lAl[n][kc * 4] = l;
    }
    // ---- stage B transposed: W_enc[k0+k][col0+m] -> lB[m][k], split hi/lo
    #pragma unroll
    for (int j = 0; j < 4; ++j) {
      int li = tid + j * 256;          // 0..1023 float4 slots
      int k  = li >> 5;                // 32 float4 per 128-col k-row
      int mc = li & 31;
      float4 wv4 = *(const float4*)&W_enc[(size_t)(k0 + k) * D_SAE + col0 + mc * 4];
      float w[4] = {wv4.x, wv4.y, wv4.z, wv4.w};
      #pragma unroll
      for (int u = 0; u < 4; ++u) {
        __bf16 h = bf_hi(w[u]);
        __bf16 l = bf_lo(w[u], h);
        lBh[mc * 4 + u][k] = h;
        lBl[mc * 4 + u][k] = l;
      }
    }
    __syncthreads();

    // ---- prefetch next chunk (global_prefetch_b8): one 128B line == one
    //      row's full 32-float chunk of x; 4 lines per W_enc k-row.
    int k1 = k0 + KC;
    if (k1 < D_MODEL) {
      __builtin_prefetch(&x[(size_t)(row0 + tid) * D_MODEL + k1], 0, 3);
      if (tid < 128) {
        int kq = tid >> 2, seg = tid & 3;
        __builtin_prefetch(&W_enc[(size_t)(k1 + kq) * D_SAE + col0 + seg * 32], 0, 3);
      }
    }

    // ---- A fragments for this wave's two 16-row bands (hi and lo)
    v16bf Ah[2], Al[2];
    #pragma unroll
    for (int rb = 0; rb < 2; ++rb) {
      int r = wv * 32 + rb * 16 + l15;
      Ah[rb] = load_frag(&lAh[r][0], lhalf);
      Al[rb] = load_frag(&lAl[r][0], lhalf);
    }

    // ---- 16 output tiles x 3 WMMAs (hi*hi + lo*hi + hi*lo), f32 accumulate
    #pragma unroll
    for (int t = 0; t < 8; ++t) {
      v16bf Bh = load_frag(&lBh[t * 16 + l15][0], lhalf);
      v16bf Bl = load_frag(&lBl[t * 16 + l15][0], lhalf);
      #pragma unroll
      for (int rb = 0; rb < 2; ++rb) {
        v8f c = acc[t][rb];
        c = __builtin_amdgcn_wmma_f32_16x16x32_bf16(false, Ah[rb], false, Bh,
                                                    (short)0, c, false, false);
        c = __builtin_amdgcn_wmma_f32_16x16x32_bf16(false, Al[rb], false, Bh,
                                                    (short)0, c, false, false);
        c = __builtin_amdgcn_wmma_f32_16x16x32_bf16(false, Ah[rb], false, Bl,
                                                    (short)0, c, false, false);
        acc[t][rb] = c;
      }
    }
    __syncthreads();
  }

  // ---- store: C/D layout -> VGPR v holds M=v (lanes 0-15) / M=v+8 (16-31)
  #pragma unroll
  for (int t = 0; t < 8; ++t) {
    int col = col0 + t * 16 + l15;
    #pragma unroll
    for (int rb = 0; rb < 2; ++rb) {
      #pragma unroll
      for (int v = 0; v < 8; ++v) {
        int row = row0 + wv * 32 + rb * 16 + v + lhalf * 8;
        pre[(size_t)row * D_SAE + col] = acc[t][rb][v];
      }
    }
  }
}

// ---------------------------------------------------------------------------
// Kernel 2: per-row top-64 via 4-level 8-bit radix select on monotonic keys.
// One 256-thread block per row; 5 streaming passes over the row (L2/HBM).
// ---------------------------------------------------------------------------
__device__ __forceinline__ unsigned int f2key(float f) {
  unsigned int b = __float_as_uint(f);
  return b ^ ((b & 0x80000000u) ? 0xFFFFFFFFu : 0x80000000u);
}

__global__ __launch_bounds__(256) void sae_topk(
    const float* __restrict__ pre, float* __restrict__ vals, int* __restrict__ idxs)
{
  __shared__ unsigned int hist[256];
  __shared__ unsigned int s_prefix, s_want, cnt_gt, cnt_eq;

  const int row = blockIdx.x;
  const int tid = threadIdx.x;
  const float* prow = pre + (size_t)row * D_SAE;

  if (tid == 0) { s_prefix = 0u; s_want = K_TOP; }
  __syncthreads();

  for (int level = 0; level < 4; ++level) {
    const int shift = 24 - 8 * level;
    hist[tid] = 0u;
    __syncthreads();
    const unsigned int prefix = s_prefix;
    const unsigned int pmask  = (level == 0) ? 0u : (0xFFFFFFFFu << (shift + 8));
    for (int i = tid; i < D_SAE; i += 256) {
      unsigned int key = f2key(prow[i]);
      if ((key & pmask) == prefix)
        atomicAdd(&hist[(key >> shift) & 0xFFu], 1u);
    }
    __syncthreads();
    if (tid == 0) {
      unsigned int want = s_want, accum = 0u;
      int b = 255;
      for (; b > 0; --b) {
        unsigned int c = hist[b];
        if (accum + c >= want) break;
        accum += c;
      }
      s_want   = want - accum;                       // ties to take at this bin
      s_prefix = s_prefix | ((unsigned int)b << shift);
    }
    __syncthreads();
  }

  const unsigned int T = s_prefix;   // full 32-bit threshold key
  const unsigned int nties = s_want; // # elements == T to take
  const unsigned int ngt = K_TOP - nties;
  if (tid == 0) { cnt_gt = 0u; cnt_eq = 0u; }
  __syncthreads();

  for (int i = tid; i < D_SAE; i += 256) {
    unsigned int key = f2key(prow[i]);
    int slot = -1;
    if (key > T) {
      slot = (int)atomicAdd(&cnt_gt, 1u);
    } else if (key == T) {
      unsigned int t = atomicAdd(&cnt_eq, 1u);
      if (t < nties) slot = (int)(ngt + t);
    }
    if (slot >= 0) {
      unsigned int b = (key & 0x80000000u) ? (key ^ 0x80000000u) : ~key;
      vals[row * K_TOP + slot] = __uint_as_float(b);
      idxs[row * K_TOP + slot] = i;
    }
  }
}

// ---------------------------------------------------------------------------
// Kernel 3: sparse decode  out[n] = b_dec + sum_k relu(vals)*W_dec[idx[k], :]
// One block per token; thread t owns cols {t*4..t*4+3} and {1024+t*4..+3}.
// ---------------------------------------------------------------------------
__global__ __launch_bounds__(256) void sae_decode(
    const float* __restrict__ vals, const int* __restrict__ idxs,
    const float* __restrict__ W_dec, const float* __restrict__ b_dec,
    float* __restrict__ out)
{
  __shared__ float sv[K_TOP];
  __shared__ int   si[K_TOP];
  const int row = blockIdx.x;
  const int tid = threadIdx.x;

  if (tid < K_TOP) {
    float v = vals[row * K_TOP + tid];
    sv[tid] = v > 0.0f ? v : 0.0f;     // ReLU folded in
    si[tid] = idxs[row * K_TOP + tid];
  }
  __syncthreads();

  float4 a0 = *(const float4*)&b_dec[tid * 4];
  float4 a1 = *(const float4*)&b_dec[1024 + tid * 4];

  for (int k = 0; k < K_TOP; ++k) {
    float v = sv[k];                   // uniform across block -> no divergence
    if (v != 0.0f) {
      const float* wr = W_dec + (size_t)si[k] * D_MODEL;
      float4 w0 = *(const float4*)&wr[tid * 4];
      float4 w1 = *(const float4*)&wr[1024 + tid * 4];
      a0.x += v * w0.x; a0.y += v * w0.y; a0.z += v * w0.z; a0.w += v * w0.w;
      a1.x += v * w1.x; a1.y += v * w1.y; a1.z += v * w1.z; a1.w += v * w1.w;
    }
  }

  *(float4*)&out[(size_t)row * D_MODEL + tid * 4] = a0;
  *(float4*)&out[(size_t)row * D_MODEL + 1024 + tid * 4] = a1;
}

// ---------------------------------------------------------------------------
extern "C" void kernel_launch(void* const* d_in, const int* in_sizes, int n_in,
                              void* d_out, int out_size, void* d_ws, size_t ws_size,
                              hipStream_t stream) {
  const float* x     = (const float*)d_in[0];
  const float* W_enc = (const float*)d_in[1];
  const float* W_dec = (const float*)d_in[2];
  const float* b_enc = (const float*)d_in[3];
  const float* b_dec = (const float*)d_in[4];
  float* out = (float*)d_out;

  // Workspace layout: pre (512 MB) | vals (2 MB) | idx (2 MB)
  float* pre  = (float*)d_ws;
  float* vals = (float*)((char*)d_ws + (size_t)N_TOK * D_SAE * sizeof(float));
  int*   idxs = (int*)((char*)vals + (size_t)N_TOK * K_TOP * sizeof(float));

  dim3 g1(D_SAE / BTN, N_TOK / BTM);   // 128 x 32 tiles
  sae_encode_gemm<<<g1, 256, 0, stream>>>(x, W_enc, b_enc, b_dec, pre);
  sae_topk<<<N_TOK, 256, 0, stream>>>(pre, vals, idxs);
  sae_decode<<<N_TOK, 256, 0, stream>>>(vals, idxs, W_dec, b_dec, out);
}
